// Attention_10213432230383
// MI455X (gfx1250) — matrix operand order repaired and checked by reading, not compiled, not run on hardware
//
#include <hip/hip_runtime.h>
#include <hip/hip_bf16.h>

typedef __attribute__((ext_vector_type(16))) _Float16 v16h;
typedef __attribute__((ext_vector_type(8)))  _Float16 v8h;
typedef __attribute__((ext_vector_type(8)))  float    v8f;
typedef __attribute__((ext_vector_type(4)))  int      v4i;

#define HQ    32   // q heads
#define HKV   8    // kv heads
#define DH    128  // head dim
#define BM    64   // q rows per block (16 per wave)
#define BN    32   // kv rows per tile
#define NWAVE 4    // waves per block (wave32)

// padded LDS rows: bank-conflict-free 16-lane fragment reads, 16B-aligned rows
#define KPAD  (DH + 8)   // f16 row = 272B -> lane bank step 4 (distinct x16)
#define VPAD  (BN + 8)   // f16 row =  80B -> lane bank step 20 (distinct x16)
#define PPAD  (16 + 8)   // f16 row =  48B -> lane bank step 12 (distinct x16)

#define ASPACE(n) __attribute__((address_space(n)))

#if defined(__gfx1250__) && __has_builtin(__builtin_amdgcn_global_load_async_to_lds_b128)
#define USE_ASYNC_LDS 1
#else
#define USE_ASYNC_LDS 0
#endif

__device__ __forceinline__ v8f wmma_f16(v16h a, v16h b, v8f c) {
  // D = A(16x32 f16) * B(32x16 f16) + C(16x16 f32)
  return __builtin_amdgcn_wmma_f32_16x16x32_f16(
      /*neg_a=*/false, a, /*neg_b=*/false, b,
      /*c_mod=*/(short)0, c, /*reuse_a=*/false, /*reuse_b=*/false);
}

__device__ __forceinline__ void copy16B_to_lds(const _Float16* g, _Float16* l) {
#if USE_ASYNC_LDS
  // per-lane 16B global -> LDS DMA, tracked by ASYNCcnt (no VGPR round trip)
  __builtin_amdgcn_global_load_async_to_lds_b128(
      (ASPACE(1) v4i*)g, (ASPACE(3) v4i*)l, 0, 0);
#else
  *(v8h*)l = *(const v8h*)g;
#endif
}

template <int N>
__device__ __forceinline__ void wait_async() {
#if USE_ASYNC_LDS
#if __has_builtin(__builtin_amdgcn_s_wait_asynccnt)
  __builtin_amdgcn_s_wait_asynccnt(N);   // immediate
#else
  asm volatile("s_wait_asynccnt %0" :: "i"(N) : "memory");
#endif
#endif
}

// ---- one-time f32 -> f16 conversion: K straight, V transposed [b][hk][d][s] ----
__global__ void convert_kv_kernel(const float* __restrict__ K,
                                  const float* __restrict__ V,
                                  const int*   __restrict__ nseq,
                                  _Float16* __restrict__ K16,
                                  _Float16* __restrict__ V16t,
                                  int T) {
  const int i = blockIdx.x * 256 + threadIdx.x;   // element in [0, T*HKV*DH)
  const int B = *nseq;
  const int S = T / B;
  K16[i] = (_Float16)K[i];
  const int t   = i / (HKV * DH);
  const int rem = i - t * (HKV * DH);
  const int hk  = rem / DH;
  const int d   = rem - hk * DH;
  const int b   = t / S;
  const int s   = t - b * S;
  V16t[(((size_t)b * HKV + hk) * DH + d) * S + s] = (_Float16)V[i];
}

template <bool PRE>
__launch_bounds__(128, 1)
__global__ void fa_fwd_kernel(const float*    __restrict__ Q,
                              const float*    __restrict__ K,
                              const float*    __restrict__ V,
                              const _Float16* __restrict__ K16,
                              const _Float16* __restrict__ V16t,
                              const int*      __restrict__ nseq,
                              float*          __restrict__ O,
                              int T) {
  __shared__ __align__(32) _Float16 Kl[BN][KPAD];            // K tile f16 (~8.7 KB)
  __shared__ __align__(32) _Float16 Vt[DH][VPAD];            // V tile f16 transposed (~10 KB)
  __shared__ __align__(32) _Float16 Pl[NWAVE][2][16][PPAD];  // per-wave P bounce (~6 KB)

  const int tid  = threadIdx.x;
  const int wave = tid >> 5;
  const int lane = tid & 31;
  const int hh   = lane >> 4;   // lane half (0/1)
  const int ln   = lane & 15;

  const int B = *nseq;          // device scalar (graph-capture safe)
  const int S = T / B;
  const int qtile = blockIdx.x;
  const int h  = blockIdx.y;
  const int hk = h / (HQ / HKV);            // GQA kv head
  const int b  = (qtile * BM) / S;
  const int q0 = qtile * BM - b * S;        // q offset within sequence
  const long seq_base = (long)b * S;
  const size_t vrow_base = ((size_t)b * HKV + hk) * DH;  // V16t [b][hk] slab

  const float scale = 0.08838834764831845f; // 1/sqrt(128)

  // ---- Q -> 4 A-fragments (16x32 f16 each), pre-scaled ----
  v16h qa[4];
  {
    const int qrow = q0 + wave * 16 + ln;
    const float* qp = Q + (((size_t)(seq_base + qrow)) * HQ + h) * DH;
    #pragma unroll
    for (int f = 0; f < 4; ++f) {
      const int kc = f * 32 + hh * 8;
      #pragma unroll
      for (int e = 0; e < 8; ++e) {
        qa[f][e]     = (_Float16)(qp[kc + e]      * scale);  // K = kc..kc+7
        qa[f][e + 8] = (_Float16)(qp[kc + 16 + e] * scale);  // K = kc+16..kc+23
      }
    }
  }

  // softmax state per C-fragment element r (row = r + 8*hh)
  float m[8], l[8];
  v8f oacc[8];
  #pragma unroll
  for (int r = 0; r < 8; ++r) { m[r] = -1e30f; l[r] = 0.f; }
  #pragma unroll
  for (int f = 0; f < 8; ++f) { v8f z = {}; oacc[f] = z; }

  const int kmax = q0 + BM;  // causal bound (exclusive)
  for (int k0 = 0; k0 < kmax; k0 += BN) {
    __syncthreads();  // previous iteration done reading tiles

    if (PRE) {
      // ---- K tile: 512 x 16B async DMA chunks (issued first) ----
      #pragma unroll
      for (int p = 0; p < 4; ++p) {
        const int c   = tid + p * 128;       // 0..511
        const int row = c >> 4;              // kv row 0..31
        const int col = (c & 15) * 8;        // f16 col 0..120
        const _Float16* g =
            K16 + ((size_t)(seq_base + k0 + row) * HKV + hk) * DH + col;
        copy16B_to_lds(g, &Kl[row][col]);
      }
      // ---- V tile: pre-transposed rows, 512 x 16B async DMA chunks ----
      #pragma unroll
      for (int p = 0; p < 4; ++p) {
        const int c   = tid + p * 128;       // 0..511
        const int d   = c >> 2;              // 0..127
        const int col = (c & 3) * 8;         // f16 col 0..24
        const _Float16* g = V16t + (vrow_base + d) * S + k0 + col;
        copy16B_to_lds(g, &Vt[d][col]);
      }
      wait_async<4>();     // K chunks (oldest) complete; V may still be in flight
      __syncthreads();     // K visible to all waves -> start QK under V DMA
    } else {
      // ---- fallback: synchronous f32->f16 fill ----
      #pragma unroll
      for (int p = 0; p < 8; ++p) {
        const int r = wave + p * 4;          // kv row 0..31
        const size_t tok = (size_t)(seq_base + k0 + r);
        const float4 kq = ((const float4*)(K + (tok * HKV + hk) * DH))[lane];
        _Float16* kd = &Kl[r][lane * 4];
        kd[0] = (_Float16)kq.x; kd[1] = (_Float16)kq.y;
        kd[2] = (_Float16)kq.z; kd[3] = (_Float16)kq.w;
        const float4 vq = ((const float4*)(V + (tok * HKV + hk) * DH))[lane];
        Vt[lane * 4 + 0][r] = (_Float16)vq.x;
        Vt[lane * 4 + 1][r] = (_Float16)vq.y;
        Vt[lane * 4 + 2][r] = (_Float16)vq.z;
        Vt[lane * 4 + 3][r] = (_Float16)vq.w;
      }
      __syncthreads();
    }

    // prefetch next kv tile (global_prefetch_b8)
    if (k0 + BN < kmax) {
      if (PRE) {
        __builtin_prefetch(K16 + ((size_t)(seq_base + k0 + BN + lane) * HKV + hk) * DH, 0, 1);
        __builtin_prefetch(V16t + (vrow_base + lane * 4) * S + k0 + BN, 0, 1);
      } else {
        __builtin_prefetch(K + ((size_t)(seq_base + k0 + BN + lane) * HKV + hk) * DH, 0, 1);
        __builtin_prefetch(V + ((size_t)(seq_base + k0 + BN + lane) * HKV + hk) * DH, 0, 1);
      }
    }

    // ---- S = Q K^T : two 16x16 score tiles per wave ----
    v8f s0 = {}, s1 = {};
    #pragma unroll
    for (int f = 0; f < 4; ++f) {
      // B fragment: lane = key column, dims 8h..8h+7 and 16+8h..16+8h+7
      const v16h b0 = __builtin_shufflevector(*(const v8h*)&Kl[ln][f * 32 + hh * 8], *(const v8h*)&Kl[ln][f * 32 + 16 + hh * 8], 0, 1, 2, 3, 4, 5, 6, 7, 8, 9, 10, 11, 12, 13, 14, 15);
      s0 = wmma_f16(qa[f], b0, s0);
      const v16h b1 = __builtin_shufflevector(*(const v8h*)&Kl[16 + ln][f * 32 + hh * 8], *(const v8h*)&Kl[16 + ln][f * 32 + 16 + hh * 8], 0, 1, 2, 3, 4, 5, 6, 7, 8, 9, 10, 11, 12, 13, 14, 15);
      s1 = wmma_f16(qa[f], b1, s1);
    }

    const bool need_mask = (k0 + BN > q0);  // wave-uniform

    // ---- online softmax (per C element r) ----
    #pragma unroll
    for (int r = 0; r < 8; ++r) {
      const int qi = q0 + wave * 16 + r + 8 * hh;
      float v0 = s0[r], v1 = s1[r];
      if (need_mask) {
        v0 = (k0 + ln      <= qi) ? v0 : -1e30f;
        v1 = (k0 + 16 + ln <= qi) ? v1 : -1e30f;
      }
      float rm = fmaxf(v0, v1);
      #pragma unroll
      for (int off = 8; off > 0; off >>= 1)   // stays within 16-lane half
        rm = fmaxf(rm, __shfl_xor(rm, off, 32));
      const float mn = fmaxf(m[r], rm);
      const float p0 = __expf(v0 - mn);
      const float p1 = __expf(v1 - mn);
      float rs = p0 + p1;
      #pragma unroll
      for (int off = 8; off > 0; off >>= 1)
        rs += __shfl_xor(rs, off, 32);
      const float alpha = __expf(m[r] - mn);
      m[r] = mn;
      l[r] = l[r] * alpha + rs;
      #pragma unroll
      for (int f = 0; f < 8; ++f) oacc[f][r] *= alpha;
      // C-layout -> LDS (row = r + 8*hh, col = ln), per-wave buffer
      Pl[wave][0][r + 8 * hh][ln] = (_Float16)p0;
      Pl[wave][1][r + 8 * hh][ln] = (_Float16)p1;
    }

    // ---- P (16x32) as A-fragment from LDS (same-wave DS is in-order) ----
    v16h pa;
    {
      const v8h plo = *(const v8h*)&Pl[wave][0][ln][8 * hh]; // K = 8h..8h+7
      const v8h phi = *(const v8h*)&Pl[wave][1][ln][8 * hh]; // K = 16+8h..16+8h+7
      #pragma unroll
      for (int e = 0; e < 8; ++e) { pa[e] = plo[e]; pa[e + 8] = phi[e]; }
    }

    if (PRE) {
      wait_async<0>();   // V DMA complete
      __syncthreads();   // V visible to all waves
    }

    // ---- O += P V : 8 d-column tiles ----
    #pragma unroll
    for (int f = 0; f < 8; ++f) {
      const v16h vb = __builtin_shufflevector(*(const v8h*)&Vt[f * 16 + ln][hh * 8], *(const v8h*)&Vt[f * 16 + ln][16 + hh * 8], 0, 1, 2, 3, 4, 5, 6, 7, 8, 9, 10, 11, 12, 13, 14, 15);
      oacc[f] = wmma_f16(pa, vb, oacc[f]);
    }
  }

  // ---- epilogue: O / l ----
  #pragma unroll
  for (int r = 0; r < 8; ++r) {
    const float inv = 1.0f / l[r];
    const int row = q0 + wave * 16 + r + 8 * hh;
    float* op = O + (((size_t)(seq_base + row)) * HQ + h) * DH;
    #pragma unroll
    for (int f = 0; f < 8; ++f)
      op[f * 16 + ln] = oacc[f][r] * inv;
  }
}

extern "C" void kernel_launch(void* const* d_in, const int* in_sizes, int n_in,
                              void* d_out, int out_size, void* d_ws, size_t ws_size,
                              hipStream_t stream) {
  (void)n_in; (void)out_size;
  const float* q    = (const float*)d_in[0];
  const float* k    = (const float*)d_in[1];
  const float* v    = (const float*)d_in[2];
  const int*   nseq = (const int*)d_in[3];
  float* out = (float*)d_out;
  const int T = in_sizes[0] / (HQ * DH);       // packed tokens
  const size_t kvElems = (size_t)T * HKV * DH; // per-tensor f16 elements
  const size_t need = 2 * kvElems * sizeof(_Float16);

  dim3 grid(T / BM, HQ, 1);                    // (64, 32) for T=4096
  dim3 block(128, 1, 1);                       // 4 waves (wave32)

  if (ws_size >= need) {
    _Float16* K16  = (_Float16*)d_ws;
    _Float16* V16t = K16 + kvElems;
    convert_kv_kernel<<<(unsigned)(kvElems / 256), 256, 0, stream>>>(
        k, v, nseq, K16, V16t, T);
    fa_fwd_kernel<true><<<grid, block, 0, stream>>>(
        q, k, v, K16, V16t, nseq, out, T);
  } else {
    fa_fwd_kernel<false><<<grid, block, 0, stream>>>(
        q, k, v, nullptr, nullptr, nseq, out, T);
  }
}
